// PhysicsLayer_76012331204759
// MI455X (gfx1250) — matrix operand order repaired and checked by reading, not compiled, not run on hardware
//
#include <hip/hip_runtime.h>
#include <hip/hip_bf16.h>
#include <math.h>

// ---------------- problem constants ----------------
#define B_      2048
#define NQ_     1024
#define WINDOW_ 51
#define HALF_   25
#define NFFT_   10000
#define HFFT_   5000
#define KPK_    8

// GEMM tiling: M (freq) padded to 320 tiles of 16 = 5120 rows; N = 2048 (128 tiles); K = 1024 (32 tiles of 32)
// One block = 4 waves; each wave owns 4 M-tiles (64 rows) x 16 batch; block = 256 rows x 16 batch.
#define MTILES_ 320
#define NTILES_ 128
#define KTILES_ 32

typedef __attribute__((ext_vector_type(16))) _Float16 v16h;
typedef __attribute__((ext_vector_type(8)))  float    v8f;
typedef __attribute__((ext_vector_type(4)))  int      v4i;

// ---------------- workspace layout (bytes) ----------------
#define WS_COEF  ((size_t)0)
#define WS_ACOS  ((size_t)16384)
#define A_ELEMS  ((size_t)MTILES_*KTILES_*512)                  // 5,242,880 f16
#define A_ALLOC  (A_ELEMS + 512)                                // +1 tile pad for unconditional kt+2 load
#define WS_ASIN  (WS_ACOS + A_ALLOC*2)
#define WS_BSW   (WS_ASIN + A_ALLOC*2)
#define B_ELEMS  ((size_t)NTILES_*KTILES_*512)                  // 2,097,152 f16
#define WS_YF    (WS_BSW + B_ELEMS*2)

// ---------------- CDNA5 async-to-LDS path (guarded) ----------------
#if __has_builtin(__builtin_amdgcn_global_load_async_to_lds_b128) && \
    __has_builtin(__builtin_amdgcn_s_wait_asynccnt)
#define USE_ASYNC_LDS 1
#else
#define USE_ASYNC_LDS 0
#endif

#define AS1 __attribute__((address_space(1)))
#define AS3 __attribute__((address_space(3)))

static __device__ __forceinline__ AS1 v4i* to_global_as(const void* p) {
  // generic global addresses are value-identical to AS1
  return (AS1 v4i*)(unsigned long long)p;
}
static __device__ __forceinline__ AS3 v4i* to_lds_as(void* p) {
  // generic LDS address: low 32 bits are the LDS byte offset (ISA aperture rule)
  return (AS3 v4i*)(unsigned int)(unsigned long long)p;
}

// =====================================================================
// Kernel 1: build Savitzky-Golay tables on device (f64 Gauss-Jordan of
// the 5x5 moment matrices; avoids baking pinv constants into source).
// =====================================================================
__global__ void sg_setup_kernel(float* coef) {
  __shared__ double Mc[5][10];
  __shared__ double Me[5][10];
  if (threadIdx.x == 0) {
    double momc[9], mome[9];
    for (int e = 0; e < 9; ++e) { momc[e] = 0.0; mome[e] = 0.0; }
    for (int x = -HALF_; x <= HALF_; ++x) { double p = 1.0; for (int e = 0; e < 9; ++e) { momc[e] += p; p *= (double)x; } }
    for (int x = 0; x < WINDOW_; ++x)     { double p = 1.0; for (int e = 0; e < 9; ++e) { mome[e] += p; p *= (double)x; } }
    for (int j = 0; j < 5; ++j)
      for (int k = 0; k < 5; ++k) {
        Mc[j][k] = momc[j + k]; Me[j][k] = mome[j + k];
        Mc[j][5 + k] = (j == k) ? 1.0 : 0.0;
        Me[j][5 + k] = (j == k) ? 1.0 : 0.0;
      }
    for (int pass = 0; pass < 2; ++pass) {
      double (*M)[10] = pass ? Me : Mc;
      for (int c = 0; c < 5; ++c) {
        int piv = c;
        for (int r = c + 1; r < 5; ++r) if (fabs(M[r][c]) > fabs(M[piv][c])) piv = r;
        if (piv != c) for (int t = 0; t < 10; ++t) { double tm = M[c][t]; M[c][t] = M[piv][t]; M[piv][t] = tm; }
        double d = M[c][c];
        for (int t = 0; t < 10; ++t) M[c][t] /= d;
        for (int r = 0; r < 5; ++r) if (r != c) {
          double fac = M[r][c];
          for (int t = 0; t < 10; ++t) M[r][t] -= fac * M[c][t];
        }
      }
    }
  }
  __syncthreads();
  for (int t = threadIdx.x; t < WINDOW_; t += blockDim.x) {
    double acc = 0.0, p = 1.0, x = (double)(t - HALF_);
    for (int k = 0; k < 5; ++k) { acc += Mc[0][5 + k] * p; p *= x; }
    coef[t] = (float)acc;
  }
  for (int idx = threadIdx.x; idx < HALF_ * WINDOW_; idx += blockDim.x) {
    int i = idx / WINDOW_, t = idx % WINDOW_;
    double accF = 0.0, accL = 0.0;
    double xiF = 1.0, xiL = 1.0;
    for (int j = 0; j < 5; ++j) {
      double xt = 1.0, rowF = 0.0;
      for (int k = 0; k < 5; ++k) { rowF += Me[j][5 + k] * xt; xt *= (double)t; }
      accF += xiF * rowF;
      accL += xiL * rowF;
      xiF *= (double)i;
      xiL *= (double)(WINDOW_ - HALF_ + i);   // 26+i
    }
    coef[WINDOW_ + idx] = (float)accF;
    coef[WINDOW_ + HALF_ * WINDOW_ + idx] = (float)accL;
  }
}

// =====================================================================
// Kernel 2: DFT basis, pre-swizzled into WMMA 16-bit A-fragment layout
// (ISA 7.12.2: lanes 0-15 hold K {0..7,16..23}, lanes 16-31 K {8..15,24..31})
// =====================================================================
__global__ void twiddle_kernel(_Float16* Acos, _Float16* Asin) {
  size_t idx = (size_t)blockIdx.x * 256 + threadIdx.x;
  if (idx >= A_ELEMS) return;
  int tile = (int)(idx >> 9);
  int within = (int)(idx & 511);
  int lane = within >> 4, slot = within & 15;
  int mt = tile >> 5, kt = tile & 31;
  int m = (lane < 16) ? lane : lane - 16;
  int kl = (lane < 16) ? ((slot < 8) ? slot : 16 + (slot - 8))
                       : ((slot < 8) ? 8 + slot : 24 + (slot - 8));
  long f = (long)mt * 16 + m;
  long n = (long)kt * 32 + kl;
  long phase = (f * n) % NFFT_;
  float ang = -6.2831853071795864f * (float)phase / (float)NFFT_;
  Acos[idx] = (_Float16)cosf(ang);
  Asin[idx] = (_Float16)sinf(ang);
}

// =====================================================================
// Kernel 3: per-sample savgol + exp10 ratio + DC removal + Hamming,
// written as f16 pre-swizzled WMMA B-fragments (lanes 0-15 K {0..7,16..23}, lanes 16-31 K {8..15,24..31}).
// =====================================================================
__global__ __launch_bounds__(256) void prep_kernel(const float* __restrict__ Rlog,
                                                   const float* __restrict__ coef,
                                                   _Float16* __restrict__ Bsw) {
  int b = blockIdx.x;
  __shared__ float y[NQ_];
  __shared__ float rs[NQ_];
  __shared__ float red[256];
  const float* yb = Rlog + (size_t)b * NQ_;
  for (int n = threadIdx.x; n < NQ_; n += 256) y[n] = yb[n];
  __syncthreads();
  const float* CC = coef;
  const float* EF = coef + WINDOW_;
  const float* EL = coef + WINDOW_ + HALF_ * WINDOW_;
  for (int i = threadIdx.x; i < NQ_; i += 256) {
    float s = 0.f;
    if (i < HALF_) {
      for (int t = 0; t < WINDOW_; ++t) s += EF[i * WINDOW_ + t] * y[t];
    } else if (i >= NQ_ - HALF_) {
      int r = i - (NQ_ - HALF_);
      for (int t = 0; t < WINDOW_; ++t) s += EL[r * WINDOW_ + t] * y[NQ_ - WINDOW_ + t];
    } else {
      for (int t = 0; t < WINDOW_; ++t) s += CC[t] * y[i - HALF_ + t];
    }
    rs[i] = exp2f((y[i] - s) * 3.3219280948873623f);   // 10^(Rlog - smooth)
  }
  __syncthreads();
  float part = 0.f;
  for (int i = threadIdx.x; i < NQ_; i += 256) part += rs[i];
  red[threadIdx.x] = part;
  __syncthreads();
  for (int st = 128; st > 0; st >>= 1) {
    if (threadIdx.x < st) red[threadIdx.x] += red[threadIdx.x + st];
    __syncthreads();
  }
  float mean = red[0] * (1.0f / NQ_);
  float hmean = 0.54f - 0.46f / (float)NQ_;   // closed-form mean(hamming)
  int nt = b >> 4, col = b & 15;
  for (int n = threadIdx.x; n < NQ_; n += 256) {
    float ham = 0.54f - 0.46f * cosf(6.2831853071795864f * (float)n / (float)(NQ_ - 1));
    float v = ham * (rs[n] - mean) / hmean;
    int kt = n >> 5, kl = n & 31;
    int lane = ((kl >> 3) & 1) ? col + 16 : col;
    int slot = (kl & 7) + ((kl >> 4) << 3);
    Bsw[((size_t)(nt * 32 + kt)) * 512 + lane * 16 + slot] = (_Float16)v;
  }
}

// =====================================================================
// Kernel 4: batched DFT as WMMA GEMM.
// 4 waves/block. The 32KB B column is staged once into LDS via
// global_load_async_to_lds_b128 (ASYNCcnt) and shared by all 4 waves.
// A fragments use a 2-deep ping-pong pipeline (kt unrolled by 2, two
// disjoint register sets -> no rotation moves, no WAR hazard nops).
// Fused magnitude on store.
// =====================================================================
__global__ __launch_bounds__(128) void dft_gemm_kernel(const _Float16* __restrict__ Acos,
                                                       const _Float16* __restrict__ Asin,
                                                       const _Float16* __restrict__ Bsw,
                                                       float* __restrict__ yf) {
  __shared__ _Float16 Bs[KTILES_ * 512];      // 32 KB
  const int nt  = blockIdx.y;
  const int tid = threadIdx.x;
  const int w   = tid >> 5;                   // wave id 0..3
  const int lane = tid & 31;
  const int mquad = blockIdx.x * 4 + w;       // 4 M-tiles per wave

  const _Float16* bcol = Bsw + (size_t)nt * KTILES_ * 512;   // contiguous 32 KB

#if USE_ASYNC_LDS
#pragma unroll
  for (int r = 0; r < 16; ++r) {
    int e = (r * 128 + tid) * 8;              // 8 f16 = 16B per lane per issue
    __builtin_amdgcn_global_load_async_to_lds_b128(
        to_global_as(bcol + e), to_lds_as(&Bs[e]), 0, 0);
  }
  __builtin_amdgcn_s_wait_asynccnt(0);
  __syncthreads();
#else
  {
    const ulonglong2* src = (const ulonglong2*)bcol;
    ulonglong2* dst = (ulonglong2*)Bs;
    for (int i = tid; i < (KTILES_ * 512) / 8; i += 128) dst[i] = src[i];
    __syncthreads();
  }
#endif

  v8f zero = {0.f, 0.f, 0.f, 0.f, 0.f, 0.f, 0.f, 0.f};
  v8f accC[4], accS[4];
#pragma unroll
  for (int mi = 0; mi < 4; ++mi) { accC[mi] = zero; accS[mi] = zero; }

  // A fragment base: tile (mt*KTILES + kt)*512 + lane*16
  const _Float16* aCbase = Acos + ((size_t)(mquad * 4) * KTILES_) * 512 + lane * 16;
  const _Float16* aSbase = Asin + ((size_t)(mquad * 4) * KTILES_) * 512 + lane * 16;

  v16h aC0[4], aS0[4], aC1[4], aS1[4];
#pragma unroll
  for (int mi = 0; mi < 4; ++mi) {
    size_t o = (size_t)mi * KTILES_ * 512;
    aC0[mi] = *(const v16h*)(aCbase + o);
    aS0[mi] = *(const v16h*)(aSbase + o);
  }

  for (int kt = 0; kt < KTILES_; kt += 2) {
    // ---- stage: load kt+1 into set1 while set0 computes kt ----
#pragma unroll
    for (int mi = 0; mi < 4; ++mi) {
      size_t o = ((size_t)mi * KTILES_ + (kt + 1)) * 512;
      aC1[mi] = *(const v16h*)(aCbase + o);
      aS1[mi] = *(const v16h*)(aSbase + o);
    }
    __builtin_prefetch(aCbase + (size_t)(kt + 2) * 512, 0, 1);
    {
      v16h bf = *(const v16h*)(&Bs[kt * 512 + lane * 16]);   // ds_load from shared B
#pragma unroll
      for (int mi = 0; mi < 4; ++mi) {
        accC[mi] = __builtin_amdgcn_wmma_f32_16x16x32_f16(false, aC0[mi], false, bf, (short)0, accC[mi], false, false);
        accS[mi] = __builtin_amdgcn_wmma_f32_16x16x32_f16(false, aS0[mi], false, bf, (short)0, accS[mi], false, false);
      }
    }
    // ---- stage: load kt+2 into set0 while set1 computes kt+1 ----
    // (kt+2 == KTILES reads the 1-tile pad; value never consumed)
#pragma unroll
    for (int mi = 0; mi < 4; ++mi) {
      size_t o = ((size_t)mi * KTILES_ + (kt + 2)) * 512;
      aC0[mi] = *(const v16h*)(aCbase + o);
      aS0[mi] = *(const v16h*)(aSbase + o);
    }
    {
      v16h bf = *(const v16h*)(&Bs[(kt + 1) * 512 + lane * 16]);
#pragma unroll
      for (int mi = 0; mi < 4; ++mi) {
        accC[mi] = __builtin_amdgcn_wmma_f32_16x16x32_f16(false, aC1[mi], false, bf, (short)0, accC[mi], false, false);
        accS[mi] = __builtin_amdgcn_wmma_f32_16x16x32_f16(false, aS1[mi], false, bf, (short)0, accS[mi], false, false);
      }
    }
  }

  // C/D layout: VGPR r, lanes 0-15 -> M=r, lanes 16-31 -> M=8+r; N = lane%16
  int col = lane & 15;
  int rofs = (lane < 16) ? 0 : 8;
  int bcolN = nt * 16 + col;
#pragma unroll
  for (int mi = 0; mi < 4; ++mi) {
    int fbase = (mquad * 4 + mi) * 16 + rofs;
#pragma unroll
    for (int r = 0; r < 8; ++r) {
      int f = fbase + r;
      if (f < HFFT_) {
        float c = accC[mi][r], s = accS[mi][r];
        yf[(size_t)f * B_ + bcolN] = (2.0f / NQ_) * sqrtf(c * c + s * s);
      }
    }
  }
}

// =====================================================================
// Kernel 5: per-sample peak detection + combination analysis.
// =====================================================================
__global__ __launch_bounds__(256) void peak_kernel(const float* __restrict__ yf,
                                                   const float* __restrict__ q,
                                                   float* __restrict__ out) {
  int b = blockIdx.x;
  __shared__ float yn[HFFT_ + 100];      // +-50 pad with -inf
  __shared__ unsigned char flags[HFFT_];
  __shared__ int cnt;
  __shared__ int ucnt;

  float dq = q[1] - q[0];
  float step = 6.2831853071795864f / ((float)NFFT_ * dq);
  float denom = yf[b] + 1e-12f;          // yf[f=0][b]

  for (int i = threadIdx.x; i < 50; i += blockDim.x) {
    yn[i] = -INFINITY;
    yn[HFFT_ + 50 + i] = -INFINITY;
  }
  if (threadIdx.x == 0) ucnt = 0;
  __syncthreads();
  int myu = 0;
  for (int f = threadIdx.x; f < HFFT_; f += blockDim.x) {
    yn[50 + f] = yf[(size_t)f * B_ + b] / denom;
    if ((float)f * step < 200.0f) myu++;
  }
  if (myu) atomicAdd(&ucnt, myu);
  __syncthreads();
  int u = ucnt;
  bool enough = u > 10;

  const int orders[4] = {50, 30, 20, 10};
  for (int oi = 0; oi < 4; ++oi) {
    int o = orders[oi];
    if (threadIdx.x == 0) cnt = 0;
    __syncthreads();
    int local = 0;
    for (int f = threadIdx.x; f < HFFT_; f += blockDim.x) {
      float v = yn[50 + f];
      bool ok = (f < u) && (v > 0.005f);
      if (ok) {
        for (int k = 1; k <= o; ++k) {
          if (!(v > yn[50 + f - k] && v > yn[50 + f + k])) { ok = false; break; }
        }
      }
      flags[f] = ok ? 1 : 0;
      if (ok) local++;
    }
    if (local) atomicAdd(&cnt, local);
    __syncthreads();
    int c = cnt;
    __syncthreads();
    if (c > 0) break;
  }
  __syncthreads();

  if (threadIdx.x == 0) {
    float p[KPK_];
    int pc = 0, tot = 0;
    float last = -INFINITY;
    if (enough) {
      for (int f = 0; f < HFFT_; ++f) {
        if (flags[f]) {
          float x = (float)f * step;
          if (x - last > 1.0f) {        // greedy min-distance, last updated on accept only
            if (pc < KPK_) p[pc++] = x;
            tot++;
            last = x;
          }
        }
      }
    }
    int count = tot < KPK_ ? tot : KPK_;
    for (int i = pc; i < KPK_; ++i) p[i] = 0.0f;

    float top3[3];
    for (int t = 0; t < 3; ++t) {
      int ts = count - 3 + t;
      ts = ts < 0 ? 0 : (ts > KPK_ - 1 ? KPK_ - 1 : ts);
      top3[t] = p[ts];
    }
    int ns = count - 3;
    float s1 = -INFINITY, s2 = -INFINITY, th1 = 0.f, th2 = 0.f;
    for (int i = 0; i < KPK_; ++i)
      for (int j = 0; j < KPK_; ++j)
        for (int t = 0; t < 3; ++t) {
          bool pok = (i <= j) && (i < ns) && (j < ns);
          float sum = p[i] + p[j];
          float err = fabsf(sum - top3[t]) / (top3[t] + 1e-12f) * 100.0f;
          bool ok = pok && (err < 10.0f) && (count >= 3);
          float score = ok ? (1.0f - err / 100.0f) : -INFINITY;
          float thick = sum * 0.5f;
          if (score > s1) { s2 = s1; th2 = th1; s1 = score; th1 = thick; }
          else if (score > s2) { s2 = score; th2 = thick; }
        }
    bool g1 = s1 > -INFINITY, g2 = s2 > -INFINITY;
    float conf = g1 ? s1 : 0.0f;
    out[b * 2 + 0] = g1 ? th1 : 0.0f;
    out[b * 2 + 1] = g2 ? th2 : 0.0f;
    out[B_ * 2 + b] = (conf > 0.1f) ? 1.0f : 0.0f;
  }
}

// =====================================================================
extern "C" void kernel_launch(void* const* d_in, const int* in_sizes, int n_in,
                              void* d_out, int out_size, void* d_ws, size_t ws_size,
                              hipStream_t stream) {
  (void)in_sizes; (void)n_in; (void)out_size; (void)ws_size;
  const float* Rlog = (const float*)d_in[0];   // [2048*1024] f32
  const float* qv   = (const float*)d_in[1];   // [1024] f32
  float* out = (float*)d_out;                  // [4096 thickness | 2048 validity]

  char* ws = (char*)d_ws;                      // needs ~66 MB
  float*     coef = (float*)(ws + WS_COEF);
  _Float16*  Acos = (_Float16*)(ws + WS_ACOS);
  _Float16*  Asin = (_Float16*)(ws + WS_ASIN);
  _Float16*  Bsw  = (_Float16*)(ws + WS_BSW);
  float*     yf   = (float*)(ws + WS_YF);

  sg_setup_kernel<<<1, 64, 0, stream>>>(coef);

  int twblocks = (int)((A_ELEMS + 255) / 256);
  twiddle_kernel<<<twblocks, 256, 0, stream>>>(Acos, Asin);

  prep_kernel<<<B_, 256, 0, stream>>>(Rlog, coef, Bsw);

  dim3 ggrid(MTILES_ / 16, NTILES_);           // 20 x 128 blocks, 128 threads (4 waves)
  dft_gemm_kernel<<<ggrid, 128, 0, stream>>>(Acos, Asin, Bsw, yf);

  peak_kernel<<<B_, 256, 0, stream>>>(yf, qv, out);
}
